// CausalTransformerDecoder_3908420239791
// MI455X (gfx1250) — compile-verified, matrix operand order checked by reading, not run on hardware
//
#include <hip/hip_runtime.h>
#include <hip/hip_bf16.h>

// Problem constants (from the reference)
#define BB    8
#define EE    512
#define TT    1024
#define CTXN  48
#define CHK   16
#define NLAY  4
#define NHEAD 8
#define FFD   2048
#define WW    64      // CTX + CHUNK
#define HDIM  64      // E / NH
#define LWIN  64      // T / CHUNK
#define NWIN  512     // B * LWIN
#define MQ    8192    // NWIN * CHK == B*T   (query/token rows)
#define MK    32768   // NWIN * WW           (key/value rows)

typedef __bf16 bhalf;
typedef __bf16 v16bh __attribute__((ext_vector_type(16)));
typedef __bf16 v8bh  __attribute__((ext_vector_type(8)));
typedef float  v8f   __attribute__((ext_vector_type(8)));

// ---------------------------------------------------------------------------
// WMMA fragment helpers (CDNA5 16x16x32 bf16 layouts, ISA 7.12.2)
// A: lane&15 = M row, lane>>4 selects K-halves {0..7,16..23} vs {8..15,24..31}
// B: identical pattern with lane&15 = N column (weight stored (N,K) row-major)
// ---------------------------------------------------------------------------
__device__ inline v16bh load_frag(const bhalf* __restrict__ base, int ld,
                                  int rbase, int kbase, int lane) {
  const int r  = rbase + (lane & 15);
  const int hi = lane >> 4;
  const bhalf* p = base + (size_t)r * ld + kbase + hi * 8;
  union { v16bh v; v8bh h[2]; } u;
  u.h[0] = *reinterpret_cast<const v8bh*>(p);        // K pairs 0..3  (k+0..7)
  u.h[1] = *reinterpret_cast<const v8bh*>(p + 16);   // K pairs 4..7  (k+16..23)
  return u.v;
}

__device__ inline v8f wmma_bf16(v16bh a, v16bh b, v8f c) {
  return __builtin_amdgcn_wmma_f32_16x16x32_bf16(false, a, false, b,
                                                 (short)0, c, false, false);
}

// ---------------------------------------------------------------------------
// Generic GEMM: C[M,N] = A[M,K](bf16) * W[N,K]^T(bf16) + bias, fused relu /
// bf16 store / transposed V store ((n,h,d,w) layout for attention B-frags).
// One wave -> 32x32 tile (2x2 fragments, 4 WMMAs / k-step => 16 FLOP/B).
// Block = 8 waves = 2(M) x 4(N) waves = 64(M) x 128(N) tile.
// bf16 weights (~33MB total) stay L2-resident (192MB); A streams once.
// ---------------------------------------------------------------------------
__global__ __launch_bounds__(256)
void gemm_bf16_kernel(const bhalf* __restrict__ A, const bhalf* __restrict__ Bw,
                      const float* __restrict__ bias,
                      float* __restrict__ Cf, bhalf* __restrict__ Cb,
                      int M, int N, int K, int relu, int vt) {
  const int lane = threadIdx.x & 31;
  const int wv   = threadIdx.x >> 5;
  const int mbase = blockIdx.y * 64  + (wv >> 2) * 32;
  const int nbase = blockIdx.x * 128 + (wv & 3) * 32;
  if (mbase >= M || nbase >= N) return;

  v8f acc00 = {}, acc01 = {}, acc10 = {}, acc11 = {};
  const bhalf* pa = A  + (size_t)(mbase + (lane & 15)) * K + (lane >> 4) * 8;
  const bhalf* pb = Bw + (size_t)(nbase + (lane & 15)) * K + (lane >> 4) * 8;

  for (int k0 = 0; k0 < K; k0 += 32) {
    // prefetch one 256B chunk ahead on both streams (global_prefetch_b8)
    __builtin_prefetch((const void*)(pa + k0 + 128), 0, 1);
    __builtin_prefetch((const void*)(pb + k0 + 128), 0, 1);

    v16bh a0 = load_frag(A,  K, mbase,      k0, lane);
    v16bh a1 = load_frag(A,  K, mbase + 16, k0, lane);
    v16bh b0 = load_frag(Bw, K, nbase,      k0, lane);
    v16bh b1 = load_frag(Bw, K, nbase + 16, k0, lane);
    acc00 = wmma_bf16(a0, b0, acc00);
    acc01 = wmma_bf16(a0, b1, acc01);
    acc10 = wmma_bf16(a1, b0, acc10);
    acc11 = wmma_bf16(a1, b1, acc11);
  }

  const int hi = lane >> 4;
  const int nl = lane & 15;
  auto store_tile = [&](v8f acc, int mb, int nb) {
    const int n = nb + nl;
    const float bs = bias ? bias[n] : 0.0f;
#pragma unroll
    for (int r = 0; r < 8; ++r) {
      const int m = mb + r + hi * 8;
      float v = acc[r] + bs;
      if (relu) v = v > 0.0f ? v : 0.0f;
      if (vt) {  // store V transposed: (win, head, d, w)
        const int nw = m >> 6, w = m & 63;
        const int h  = n >> 6, d = n & 63;
        Cb[(((size_t)nw * NHEAD + h) * HDIM + d) * WW + w] = (bhalf)v;
      } else {
        if (Cf) Cf[(size_t)m * N + n] = v;
        if (Cb) Cb[(size_t)m * N + n] = (bhalf)v;
      }
    }
  };
  store_tile(acc00, mbase,      nbase);
  store_tile(acc01, mbase,      nbase + 16);
  store_tile(acc10, mbase + 16, nbase);
  store_tile(acc11, mbase + 16, nbase + 16);
}

// ---------------------------------------------------------------------------
// Attention: one wave per (window n, head h).  Tq=16, Tk=64, HD=64.
// scores = (Q Kt)/8 -> softmax (rows via lanes 0..15 through LDS) -> P V.
// ---------------------------------------------------------------------------
__global__ __launch_bounds__(256)
void attn_kernel(const bhalf* __restrict__ qp, const bhalf* __restrict__ kp,
                 const bhalf* __restrict__ vpT, bhalf* __restrict__ o) {
  __shared__ __align__(16) float sS[8][16][68];
  __shared__ __align__(16) bhalf sP[8][16][64];

  const int lane = threadIdx.x & 31;
  const int wv   = threadIdx.x >> 5;
  const int idx  = blockIdx.x * 8 + wv;
  const int n = idx >> 3, h = idx & 7;

  const bhalf* qbase = qp  + ((size_t)n * CHK) * EE + h * HDIM;
  const bhalf* kbase = kp  + ((size_t)n * WW)  * EE + h * HDIM;
  const bhalf* vbase = vpT + ((size_t)(n * NHEAD + h)) * HDIM * WW;

  const int hi = lane >> 4, nl = lane & 15;

  // ---- scores: Q(16x64) @ K(64x64)^T, key tiles t = 0..3 ----
  v16bh aq0 = load_frag(qbase, EE, 0, 0,  lane);
  v16bh aq1 = load_frag(qbase, EE, 0, 32, lane);
#pragma unroll
  for (int t = 0; t < 4; ++t) {
    v8f acc = {};
    v16bh b0 = load_frag(kbase, EE, t * 16, 0,  lane);
    v16bh b1 = load_frag(kbase, EE, t * 16, 32, lane);
    acc = wmma_bf16(aq0, b0, acc);
    acc = wmma_bf16(aq1, b1, acc);
#pragma unroll
    for (int r = 0; r < 8; ++r)
      sS[wv][r + hi * 8][t * 16 + nl] = acc[r] * 0.125f;  // 1/sqrt(64)
  }
  __syncthreads();

  // ---- softmax: lanes 0..15 each own one query row ----
  if (lane < 16) {
    float mx = -3.4e38f;
    for (int j = 0; j < WW; ++j) mx = fmaxf(mx, sS[wv][lane][j]);
    float sum = 0.0f;
    for (int j = 0; j < WW; ++j) {
      float ev = __expf(sS[wv][lane][j] - mx);
      sum += ev;
      sS[wv][lane][j] = ev;
    }
    const float inv = 1.0f / sum;
    for (int j = 0; j < WW; ++j)
      sP[wv][lane][j] = (bhalf)(sS[wv][lane][j] * inv);
  }
  __syncthreads();

  // ---- out: P(16x64) @ V(64x64), output d-tiles t = 0..3 ----
  v16bh p0 = load_frag(&sP[wv][0][0], WW, 0, 0,  lane);
  v16bh p1 = load_frag(&sP[wv][0][0], WW, 0, 32, lane);
#pragma unroll
  for (int t = 0; t < 4; ++t) {
    v8f acc = {};
    v16bh b0 = load_frag(vbase, WW, t * 16, 0,  lane);
    v16bh b1 = load_frag(vbase, WW, t * 16, 32, lane);
    acc = wmma_bf16(p0, b0, acc);
    acc = wmma_bf16(p1, b1, acc);
#pragma unroll
    for (int r = 0; r < 8; ++r)
      o[(size_t)(n * CHK + r + hi * 8) * EE + h * HDIM + t * 16 + nl] =
          (bhalf)acc[r];
  }
}

// ---------------------------------------------------------------------------
// Residual + LayerNorm: one wave per row of 512; wave32 shfl reductions.
// ---------------------------------------------------------------------------
__global__ __launch_bounds__(256)
void residual_ln_kernel(float* __restrict__ x, bhalf* __restrict__ xb,
                        const float* __restrict__ t,
                        const float* __restrict__ g,
                        const float* __restrict__ b) {
  const int lane = threadIdx.x & 31;
  const size_t row = (size_t)blockIdx.x * 8 + (threadIdx.x >> 5);
  const float* xr = x + row * EE;
  const float* tr = t + row * EE;

  float v[16], s = 0.0f, sq = 0.0f;
#pragma unroll
  for (int j = 0; j < 16; ++j) {
    const float u = xr[lane + j * 32] + tr[lane + j * 32];
    v[j] = u; s += u; sq += u * u;
  }
#pragma unroll
  for (int off = 16; off >= 1; off >>= 1) {
    s  += __shfl_xor(s,  off, 32);
    sq += __shfl_xor(sq, off, 32);
  }
  const float mean = s * (1.0f / EE);
  const float var  = sq * (1.0f / EE) - mean * mean;
  const float inv  = rsqrtf(var + 1e-5f);
#pragma unroll
  for (int j = 0; j < 16; ++j) {
    const int e = lane + j * 32;
    const float y = (v[j] - mean) * inv * g[e] + b[e];
    x[row * EE + e]  = y;
    xb[row * EE + e] = (bhalf)y;
  }
}

// ---------------------------------------------------------------------------
// Elementwise / layout kernels
// ---------------------------------------------------------------------------
__global__ __launch_bounds__(256)
void cvt_f32_bf16_kernel(const float* __restrict__ s, bhalf* __restrict__ d, int n) {
  const int i = blockIdx.x * 256 + threadIdx.x;
  if (i < n) d[i] = (bhalf)s[i];
}

// x[b,t,e] = tgt[b,e,t]
__global__ __launch_bounds__(256)
void init_x_kernel(const float* __restrict__ tgt, float* __restrict__ x,
                   bhalf* __restrict__ xb) {
  const int i = blockIdx.x * 256 + threadIdx.x;       // < B*T*E
  const int e = i & 511, bt = i >> 9, t = bt & 1023, b = bt >> 10;
  const float v = tgt[((size_t)(b * EE + e)) * TT + t];
  x[i] = v; xb[i] = (bhalf)v;
}

// mem_ctx_bf[n,w,e] = unfold(concat(ctx_buf[:,0], mem^T))[n,w,e] + pe[w,e]
__global__ __launch_bounds__(256)
void build_mem_ctx_kernel(const float* __restrict__ mem,
                          const float* __restrict__ ctx,
                          const float* __restrict__ pe,
                          bhalf* __restrict__ mc) {
  const int i = blockIdx.x * 256 + threadIdx.x;       // < NWIN*WW*E
  const int e = i & 511, nw = i >> 9, w = nw & 63, n = nw >> 6;
  const int b = n >> 6, l = n & 63, t = l * CHK + w;
  const float mv = (t < CTXN)
      ? ctx[((size_t)(b * (NLAY + 1)) * CTXN + t) * EE + e]
      : mem[((size_t)(b * EE + e)) * TT + (t - CTXN)];
  mc[i] = (bhalf)(mv + pe[w * EE + e]);
}

// tc_bf[n,w,e] = unfold(concat(ctx_buf[:,li+1], x))[n,w,e] (+ pe on layer 0)
__global__ __launch_bounds__(256)
void build_tc_kernel(const float* __restrict__ x, const float* __restrict__ ctx,
                     const float* __restrict__ pe, bhalf* __restrict__ tc,
                     int li) {
  const int i = blockIdx.x * 256 + threadIdx.x;
  const int e = i & 511, nw = i >> 9, w = nw & 63, n = nw >> 6;
  const int b = n >> 6, l = n & 63, t = l * CHK + w;
  float v = (t < CTXN)
      ? ctx[(((size_t)b * (NLAY + 1) + (li + 1)) * CTXN + t) * EE + e]
      : x[((size_t)b * TT + (t - CTXN)) * EE + e];
  if (li == 0) v += pe[w * EE + e];
  tc[i] = (bhalf)v;
}

// ctx_out[:,0] = tail of mem stream
__global__ __launch_bounds__(256)
void write_ctx_mem_kernel(const float* __restrict__ mem, float* __restrict__ co) {
  const int i = blockIdx.x * 256 + threadIdx.x;       // < B*CTX*E
  const int e = i & 511, bc = i >> 9, c = bc % CTXN, b = bc / CTXN;
  co[(((size_t)b * (NLAY + 1)) * CTXN + c) * EE + e] =
      mem[((size_t)(b * EE + e)) * TT + (TT - CTXN + c)];
}

// ctx_out[:,li+1] = tail of current x (pre-layer)
__global__ __launch_bounds__(256)
void write_ctx_layer_kernel(const float* __restrict__ x, float* __restrict__ co,
                            int li) {
  const int i = blockIdx.x * 256 + threadIdx.x;
  const int e = i & 511, bc = i >> 9, c = bc % CTXN, b = bc / CTXN;
  co[(((size_t)b * (NLAY + 1) + (li + 1)) * CTXN + c) * EE + e] =
      x[((size_t)b * TT + (TT - CTXN + c)) * EE + e];
}

// out[b,e,t] = x[b,t,e]
__global__ __launch_bounds__(256)
void write_out_kernel(const float* __restrict__ x, float* __restrict__ out) {
  const int i = blockIdx.x * 256 + threadIdx.x;
  const int e = i & 511, bt = i >> 9, t = bt & 1023, b = bt >> 10;
  out[((size_t)(b * EE + e)) * TT + t] = x[i];
}

// ---------------------------------------------------------------------------
// Host orchestration
// ---------------------------------------------------------------------------
extern "C" void kernel_launch(void* const* d_in, const int* in_sizes, int n_in,
                              void* d_out, int out_size, void* d_ws, size_t ws_size,
                              hipStream_t stream) {
  (void)in_sizes; (void)n_in; (void)out_size; (void)ws_size;

  const float* tgt   = (const float*)d_in[0];
  const float* mem   = (const float*)d_in[1];
  const float* ctxb  = (const float*)d_in[2];
  const float* pe    = (const float*)d_in[3];
  const float* sa_w  = (const float*)d_in[4];
  const float* sa_b  = (const float*)d_in[5];
  const float* sa_ow = (const float*)d_in[6];
  const float* sa_ob = (const float*)d_in[7];
  const float* ca_w  = (const float*)d_in[8];
  const float* ca_b  = (const float*)d_in[9];
  const float* ca_ow = (const float*)d_in[10];
  const float* ca_ob = (const float*)d_in[11];
  const float* l1_w  = (const float*)d_in[12];
  const float* l1_b  = (const float*)d_in[13];
  const float* l2_w  = (const float*)d_in[14];
  const float* l2_b  = (const float*)d_in[15];
  const float* n1_g  = (const float*)d_in[16];
  const float* n1_b  = (const float*)d_in[17];
  const float* n2_g  = (const float*)d_in[18];
  const float* n2_b  = (const float*)d_in[19];
  const float* n3_g  = (const float*)d_in[20];
  const float* n3_b  = (const float*)d_in[21];

  float* out_x  = (float*)d_out;                        // (B,E,T)
  float* out_cx = (float*)d_out + (size_t)BB * EE * TT; // (B,NL+1,CTX,E)

  // workspace carve-up (256B aligned)
  char* ws = (char*)d_ws;
  size_t off = 0;
  auto carve = [&](size_t bytes) {
    char* p = ws + off;
    off += (bytes + 255) & ~(size_t)255;
    return p;
  };
  bhalf* sa_w_bf  = (bhalf*)carve((size_t)NLAY * 3 * EE * EE * 2);
  bhalf* ca_w_bf  = (bhalf*)carve((size_t)NLAY * 3 * EE * EE * 2);
  bhalf* sa_ow_bf = (bhalf*)carve((size_t)NLAY * EE * EE * 2);
  bhalf* ca_ow_bf = (bhalf*)carve((size_t)NLAY * EE * EE * 2);
  bhalf* l1_bf    = (bhalf*)carve((size_t)NLAY * FFD * EE * 2);
  bhalf* l2_bf    = (bhalf*)carve((size_t)NLAY * EE * FFD * 2);
  bhalf* mc_bf    = (bhalf*)carve((size_t)MK * EE * 2);
  float* x_f      = (float*)carve((size_t)MQ * EE * 4);
  bhalf* x_bf     = (bhalf*)carve((size_t)MQ * EE * 2);
  bhalf* tc_bf    = (bhalf*)carve((size_t)MK * EE * 2);
  bhalf* qp_bf    = (bhalf*)carve((size_t)MQ * EE * 2);
  bhalf* kp_bf    = (bhalf*)carve((size_t)MK * EE * 2);
  bhalf* vpT_bf   = (bhalf*)carve((size_t)MK * EE * 2);
  bhalf* o_bf     = (bhalf*)carve((size_t)MQ * EE * 2);
  float* tmp_f    = (float*)carve((size_t)MQ * EE * 4);
  bhalf* h_bf     = (bhalf*)carve((size_t)MQ * FFD * 2);

  auto cvt = [&](const float* s, bhalf* d, int n) {
    cvt_f32_bf16_kernel<<<(n + 255) / 256, 256, 0, stream>>>(s, d, n);
  };
  auto gemm = [&](const bhalf* A, const bhalf* Bw, const float* bias,
                  float* Cf, bhalf* Cb, int M, int N, int K, int relu, int vt) {
    gemm_bf16_kernel<<<dim3(N / 128, M / 64), 256, 0, stream>>>(
        A, Bw, bias, Cf, Cb, M, N, K, relu, vt);
  };

  // weight conversions (fp32 -> bf16, (N,K) row-major == B-fragment layout)
  cvt(sa_w,  sa_w_bf,  NLAY * 3 * EE * EE);
  cvt(ca_w,  ca_w_bf,  NLAY * 3 * EE * EE);
  cvt(sa_ow, sa_ow_bf, NLAY * EE * EE);
  cvt(ca_ow, ca_ow_bf, NLAY * EE * EE);
  cvt(l1_w,  l1_bf,    NLAY * FFD * EE);
  cvt(l2_w,  l2_bf,    NLAY * EE * FFD);

  const int NX  = MQ * EE;        // 4,194,304
  const int NTC = MK * EE;        // 16,777,216
  const int NCX = BB * CTXN * EE; // 196,608
  init_x_kernel<<<NX / 256, 256, 0, stream>>>(tgt, x_f, x_bf);
  build_mem_ctx_kernel<<<NTC / 256, 256, 0, stream>>>(mem, ctxb, pe, mc_bf);
  write_ctx_mem_kernel<<<NCX / 256, 256, 0, stream>>>(mem, out_cx);

  for (int i = 0; i < NLAY; ++i) {
    write_ctx_layer_kernel<<<NCX / 256, 256, 0, stream>>>(x_f, out_cx, i);
    build_tc_kernel<<<NTC / 256, 256, 0, stream>>>(x_f, ctxb, pe, tc_bf, i);

    const bhalf* saw = sa_w_bf + (size_t)i * 3 * EE * EE;
    const bhalf* caw = ca_w_bf + (size_t)i * 3 * EE * EE;

    // ---- self-attention ----
    gemm(x_bf,  saw,               sa_b + i * 3 * EE,          nullptr, qp_bf,  MQ, EE, EE, 0, 0);
    gemm(tc_bf, saw + EE * EE,     sa_b + i * 3 * EE + EE,     nullptr, kp_bf,  MK, EE, EE, 0, 0);
    gemm(tc_bf, saw + 2 * EE * EE, sa_b + i * 3 * EE + 2 * EE, nullptr, vpT_bf, MK, EE, EE, 0, 1);
    attn_kernel<<<NWIN * NHEAD / 8, 256, 0, stream>>>(qp_bf, kp_bf, vpT_bf, o_bf);
    gemm(o_bf, sa_ow_bf + (size_t)i * EE * EE, sa_ob + i * EE, tmp_f, nullptr, MQ, EE, EE, 0, 0);
    residual_ln_kernel<<<MQ / 8, 256, 0, stream>>>(x_f, x_bf, tmp_f,
                                                   n1_g + i * EE, n1_b + i * EE);

    // ---- cross-attention (kv from mem_ctx) ----
    gemm(x_bf,  caw,               ca_b + i * 3 * EE,          nullptr, qp_bf,  MQ, EE, EE, 0, 0);
    gemm(mc_bf, caw + EE * EE,     ca_b + i * 3 * EE + EE,     nullptr, kp_bf,  MK, EE, EE, 0, 0);
    gemm(mc_bf, caw + 2 * EE * EE, ca_b + i * 3 * EE + 2 * EE, nullptr, vpT_bf, MK, EE, EE, 0, 1);
    attn_kernel<<<NWIN * NHEAD / 8, 256, 0, stream>>>(qp_bf, kp_bf, vpT_bf, o_bf);
    gemm(o_bf, ca_ow_bf + (size_t)i * EE * EE, ca_ob + i * EE, tmp_f, nullptr, MQ, EE, EE, 0, 0);
    residual_ln_kernel<<<MQ / 8, 256, 0, stream>>>(x_f, x_bf, tmp_f,
                                                   n2_g + i * EE, n2_b + i * EE);

    // ---- feed-forward ----
    gemm(x_bf, l1_bf + (size_t)i * FFD * EE, l1_b + i * FFD, nullptr, h_bf, MQ, FFD, EE, 1, 0);
    gemm(h_bf, l2_bf + (size_t)i * EE * FFD, l2_b + i * EE,  tmp_f, nullptr, MQ, EE, FFD, 0, 0);
    residual_ln_kernel<<<MQ / 8, 256, 0, stream>>>(x_f, x_bf, tmp_f,
                                                   n3_g + i * EE, n3_b + i * EE);
  }

  write_out_kernel<<<NX / 256, 256, 0, stream>>>(x_f, out_x);
}
